// ExpertParallelMoE_73512660238766
// MI455X (gfx1250) — matrix operand order repaired and checked by reading, not compiled, not run on hardware
//
#include <hip/hip_runtime.h>
#include <hip/hip_bf16.h>

#define N_EXPERTS 8
#define D_MODEL   1024
#define D_FF      4096
#define N_TOKENS  2048
#define MROWS     64                      // token rows per block (4 WMMA sub-tiles)
#define MAX_MBLKS (N_TOKENS / MROWS)      // 32

typedef __bf16 v16bf __attribute__((ext_vector_type(16)));
typedef float  v8f   __attribute__((ext_vector_type(8)));

// ---------------------------------------------------------------------------
// Kernel 1: per-expert token compaction (counts + permutation lists)
// ---------------------------------------------------------------------------
__global__ void moe_route_kernel(const int* __restrict__ expert_idx,
                                 int* __restrict__ counts,
                                 int* __restrict__ perm) {
    int t = blockIdx.x * blockDim.x + threadIdx.x;
    if (t >= N_TOKENS) return;
    int e = expert_idx[t];
    int slot = atomicAdd(&counts[e], 1);
    perm[e * N_TOKENS + slot] = t;
}

// ---------------------------------------------------------------------------
// Kernel 2: h = silu(X_e @ Wg) * (X_e @ Wu).
// Block = 128 threads (4 waves) computes a 64x64 strip of h for one expert.
// Each wave owns a 16-column strip; B fragments (gate+up) are loaded once per
// k-step and reused across 4 M sub-tiles -> 8 WMMAs per 32 B-loads.
// A-staging is branchless: padding rows read a valid (clamped) token row;
// garbage there only affects output rows that are never stored.
// ---------------------------------------------------------------------------
__global__ __launch_bounds__(128, 1)
void moe_gateup_kernel(const float* __restrict__ x,
                       const float* __restrict__ w_gate,
                       const float* __restrict__ w_up,
                       const int*   __restrict__ counts,
                       const int*   __restrict__ perm,
                       __bf16*      __restrict__ h) {
    const int e     = blockIdx.z;
    const int cnt   = counts[e];
    const int mbase = blockIdx.y * MROWS;
    if (mbase >= cnt) return;                       // block-uniform exit

    const int wave  = threadIdx.x >> 5;             // 0..3
    const int lane  = threadIdx.x & 31;
    const int half  = lane >> 4;                    // 0/1 lane half
    const int ln    = lane & 15;                    // row (A) / col (B,C)
    const int nbase = blockIdx.x * 64 + wave * 16;  // column base in D_FF

    __shared__ __bf16 Atile[MROWS][32];
    __shared__ int    rows[MROWS];

    if (threadIdx.x < MROWS) {
        int r = mbase + threadIdx.x;
        rows[threadIdx.x] = (r < cnt) ? perm[e * N_TOKENS + r] : -1;
    }
    __syncthreads();

    // Per-thread staging pointers, computed once (branchless inner loop).
    const float* asrc[4];
    __bf16*      adst[4];
#pragma unroll
    for (int s = 0; s < 4; ++s) {
        const int linear = threadIdx.x + s * 128;  // float4 slot 0..511
        const int arow   = linear >> 3;            // 8 float4 per row
        const int acol   = (linear & 7) * 4;
        int trow = rows[arow];
        if (trow < 0) trow = rows[0];              // rows[0] always valid
        asrc[s] = x + (size_t)trow * D_MODEL + acol;
        adst[s] = &Atile[arow][acol];
    }

    const size_t woff = (size_t)e * D_MODEL * D_FF;
    const float* wg = w_gate + woff + nbase + ln;
    const float* wu = w_up   + woff + nbase + ln;

    v8f zero = {0.f, 0.f, 0.f, 0.f, 0.f, 0.f, 0.f, 0.f};
    v8f accg[4], accu[4];
#pragma unroll
    for (int s = 0; s < 4; ++s) { accg[s] = zero; accu[s] = zero; }

    for (int k0 = 0; k0 < D_MODEL; k0 += 32) {
        // Stage gathered A tile (64 x 32 f32 -> bf16) into LDS, coalesced,
        // no branches: 4 batched b128 loads per thread.
#pragma unroll
        for (int s = 0; s < 4; ++s) {
            const float4 v = *(const float4*)(asrc[s] + k0);
            adst[s][0] = (__bf16)v.x;
            adst[s][1] = (__bf16)v.y;
            adst[s][2] = (__bf16)v.z;
            adst[s][3] = (__bf16)v.w;
        }
        __syncthreads();

        // B fragments: lane ln = column N; elem j -> K = half*16 + j.
        float fg[16], fu[16];
#pragma unroll
        for (int j = 0; j < 16; ++j) {
            const size_t krow = (size_t)(k0 + half * 16 + j) * D_FF;
            fg[j] = wg[krow];
            fu[j] = wu[krow];
        }
        v16bf bg, bu;
#pragma unroll
        for (int j = 0; j < 16; ++j) {
            bg[j] = (__bf16)fg[j];
            bu[j] = (__bf16)fu[j];
        }

        // 4 M sub-tiles reuse bg/bu: 8 WMMAs per k-step.
#pragma unroll
        for (int sub = 0; sub < 4; ++sub) {
            v16bf a;
#pragma unroll
            for (int j = 0; j < 8; ++j) {
                a[j]     = Atile[sub * 16 + ln][half * 16 + j];
                a[j + 8] = Atile[sub * 16 + ln][half * 16 + 8 + j];
            }
            accg[sub] = __builtin_amdgcn_wmma_f32_16x16x32_bf16(false, a, false, bg,
                                                                (short)0, accg[sub], false, false);
            accu[sub] = __builtin_amdgcn_wmma_f32_16x16x32_bf16(false, a, false, bu,
                                                                (short)0, accu[sub], false, false);
        }
        __syncthreads();
    }

    // Epilogue: h = silu(gate) * up, scattered by token id (bf16, L2-resident)
#pragma unroll
    for (int sub = 0; sub < 4; ++sub) {
#pragma unroll
        for (int r = 0; r < 8; ++r) {
            const int M = sub * 16 + r + 8 * half;    // C/D layout: M = r + 8*half
            const int trow = rows[M];
            if (trow >= 0) {
                float g = accg[sub][r];
                float u = accu[sub][r];
                float s = g / (1.0f + __expf(-g));    // silu
                h[(size_t)trow * D_FF + nbase + ln] = (__bf16)(s * u);
            }
        }
    }
}

// ---------------------------------------------------------------------------
// Kernel 3: out = h @ Wd (K = 4096), f32 output scattered by token id.
// Same 64-row blocking: 4 WMMAs per 16 B-loads; branchless A-staging.
// ---------------------------------------------------------------------------
__global__ __launch_bounds__(128, 1)
void moe_down_kernel(const __bf16* __restrict__ h,
                     const float*  __restrict__ w_down,
                     const int*    __restrict__ counts,
                     const int*    __restrict__ perm,
                     float*        __restrict__ out) {
    const int e     = blockIdx.z;
    const int cnt   = counts[e];
    const int mbase = blockIdx.y * MROWS;
    if (mbase >= cnt) return;

    const int wave  = threadIdx.x >> 5;
    const int lane  = threadIdx.x & 31;
    const int half  = lane >> 4;
    const int ln    = lane & 15;
    const int nbase = blockIdx.x * 64 + wave * 16;  // column base in D_MODEL

    __shared__ __bf16 Atile[MROWS][32];
    __shared__ int    rows[MROWS];

    if (threadIdx.x < MROWS) {
        int r = mbase + threadIdx.x;
        rows[threadIdx.x] = (r < cnt) ? perm[e * N_TOKENS + r] : -1;
    }
    __syncthreads();

    const __bf16* asrc[4];
    __bf16*       adst[4];
#pragma unroll
    for (int s = 0; s < 4; ++s) {
        const int linear = threadIdx.x + s * 128;  // 4-elem slot 0..511
        const int arow   = linear >> 3;
        const int acol   = (linear & 7) * 4;
        int trow = rows[arow];
        if (trow < 0) trow = rows[0];
        asrc[s] = h + (size_t)trow * D_FF + acol;
        adst[s] = &Atile[arow][acol];
    }

    const float* wd = w_down + (size_t)e * D_FF * D_MODEL + nbase + ln;

    v8f zero = {0.f, 0.f, 0.f, 0.f, 0.f, 0.f, 0.f, 0.f};
    v8f acc[4];
#pragma unroll
    for (int s = 0; s < 4; ++s) acc[s] = zero;

    for (int k0 = 0; k0 < D_FF; k0 += 32) {
        // Stage gathered A tile (64 x 32 bf16) into LDS: 8B chunks, branchless.
#pragma unroll
        for (int s = 0; s < 4; ++s) {
            const __bf16* src = asrc[s] + k0;
            adst[s][0] = src[0];
            adst[s][1] = src[1];
            adst[s][2] = src[2];
            adst[s][3] = src[3];
        }
        __syncthreads();

        float fb[16];
#pragma unroll
        for (int j = 0; j < 16; ++j) {
            const size_t krow = (size_t)(k0 + half * 16 + j) * D_MODEL;
            fb[j] = wd[krow];
        }
        v16bf b;
#pragma unroll
        for (int j = 0; j < 16; ++j) b[j] = (__bf16)fb[j];

#pragma unroll
        for (int sub = 0; sub < 4; ++sub) {
            v16bf a;
#pragma unroll
            for (int j = 0; j < 8; ++j) {
                a[j]     = Atile[sub * 16 + ln][half * 16 + j];
                a[j + 8] = Atile[sub * 16 + ln][half * 16 + 8 + j];
            }
            acc[sub] = __builtin_amdgcn_wmma_f32_16x16x32_bf16(false, a, false, b,
                                                               (short)0, acc[sub], false, false);
        }
        __syncthreads();
    }

#pragma unroll
    for (int sub = 0; sub < 4; ++sub) {
#pragma unroll
        for (int r = 0; r < 8; ++r) {
            const int M = sub * 16 + r + 8 * half;
            const int trow = rows[M];
            if (trow >= 0) {
                out[(size_t)trow * D_MODEL + nbase + ln] = acc[sub][r];
            }
        }
    }
}

// ---------------------------------------------------------------------------
// Host launcher
// ---------------------------------------------------------------------------
extern "C" void kernel_launch(void* const* d_in, const int* in_sizes, int n_in,
                              void* d_out, int out_size, void* d_ws, size_t ws_size,
                              hipStream_t stream) {
    const float* x          = (const float*)d_in[0];
    const int*   expert_idx = (const int*)d_in[1];
    const float* w_gate     = (const float*)d_in[2];
    const float* w_up       = (const float*)d_in[3];
    const float* w_down     = (const float*)d_in[4];
    float*       out        = (float*)d_out;

    char* ws = (char*)d_ws;
    int*    counts = (int*)(ws + 0);                         // 8 ints
    int*    perm   = (int*)(ws + 256);                       // 8*2048 ints = 64KB
    __bf16* h      = (__bf16*)(ws + 256 + 65536);            // 2048*4096 bf16 = 16MB

    hipMemsetAsync(counts, 0, N_EXPERTS * sizeof(int), stream);

    moe_route_kernel<<<(N_TOKENS + 255) / 256, 256, 0, stream>>>(expert_idx, counts, perm);

    dim3 gridGU(D_FF / 64, MAX_MBLKS, N_EXPERTS);
    moe_gateup_kernel<<<gridGU, 128, 0, stream>>>(x, w_gate, w_up, counts, perm, h);

    dim3 gridDN(D_MODEL / 64, MAX_MBLKS, N_EXPERTS);
    moe_down_kernel<<<gridDN, 128, 0, stream>>>(h, w_down, counts, perm, out);
}
